// GlobalPointer_27633819582893
// MI455X (gfx1250) — compile-verified, matrix operand order checked by reading, not run on hardware
//
#include <hip/hip_runtime.h>
#include <hip/hip_bf16.h>
#include <cstdint>
#include <cstddef>

// ---------- CDNA5 WMMA types ----------
typedef __attribute__((ext_vector_type(16))) __bf16 v16bf;
typedef __attribute__((ext_vector_type(8)))  float  v8f;
typedef __attribute__((ext_vector_type(4)))  int    v4i;

union BF16x16 { v16bf v; __bf16 e[16]; };
union F32x8   { v8f   v; float  e[8];  };

#define NB    8        // batch
#define SEQ   512
#define HID   768
#define GHID  384      // GRU hidden
#define KDIM  1536     // dense K
#define NOUT  6656     // 52 * 128
#define NENT  52
#define DHEAD 64
#define BIGF  1000000000000.0f
// ln(10000)/32
#define LOG1E4_OVER_32 0.28782313662425576f

#if defined(__HIP_DEVICE_COMPILE__) && __has_builtin(__builtin_amdgcn_global_load_async_to_lds_b128)
#define HAVE_ASYNC_LDS 1
#else
#define HAVE_ASYNC_LDS 0
#endif

__device__ __forceinline__ void wait_async_zero() {
#if defined(__HIP_DEVICE_COMPILE__)
#if __has_builtin(__builtin_amdgcn_s_wait_asynccnt)
    __builtin_amdgcn_s_wait_asynccnt(0);
#else
    asm volatile("s_wait_asynccnt 0" ::: "memory");
#endif
#endif
}

// =====================================================================
// Kernel A: two single-step GRU cells -> cls_emb (8 x 768) f32
// =====================================================================
__global__ __launch_bounds__(256) void gru_kernel(
    const float* __restrict__ cls_pooled,  // (8,768)
    const float* __restrict__ h0,          // (2,8,384)
    const float* __restrict__ w_ih_f, const float* __restrict__ w_hh_f,
    const float* __restrict__ b_ih_f, const float* __restrict__ b_hh_f,
    const float* __restrict__ w_ih_b, const float* __restrict__ w_hh_b,
    const float* __restrict__ b_ih_b, const float* __restrict__ b_hh_b,
    float* __restrict__ cls_emb)           // (8,768): [fwd|bwd]
{
    int gid = blockIdx.x * blockDim.x + threadIdx.x;
    if (gid >= 2 * NB * GHID) return;
    int dir = gid / (NB * GHID);
    int rem = gid % (NB * GHID);
    int b   = rem / GHID;
    int j   = rem % GHID;

    const float* w_ih = dir ? w_ih_b : w_ih_f;
    const float* w_hh = dir ? w_hh_b : w_hh_f;
    const float* b_ih = dir ? b_ih_b : b_ih_f;
    const float* b_hh = dir ? b_hh_b : b_hh_f;

    const float* x = cls_pooled + (size_t)b * HID;
    const float* h = h0 + ((size_t)dir * NB + b) * GHID;

    float gi[3], gh[3];
    #pragma unroll
    for (int g = 0; g < 3; ++g) {
        int row = g * GHID + j;
        float acc = b_ih[row];
        const float* wr = w_ih + (size_t)row * HID;
        for (int k = 0; k < HID; ++k) acc += wr[k] * x[k];
        gi[g] = acc;
        acc = b_hh[row];
        const float* wr2 = w_hh + (size_t)row * GHID;
        for (int k = 0; k < GHID; ++k) acc += wr2[k] * h[k];
        gh[g] = acc;
    }
    float r = 1.0f / (1.0f + __expf(-(gi[0] + gh[0])));
    float z = 1.0f / (1.0f + __expf(-(gi[1] + gh[1])));
    float n = tanhf(gi[2] + r * gh[2]);
    cls_emb[(size_t)b * HID + dir * GHID + j] = (1.0f - z) * n + z * h[j];
}

// =====================================================================
// Kernel B: out = [lhs | cls_emb] @ dense_w.T + dense_b, fused RoPE,
//           writes qw/kw as bf16 in (b,ent,s,d) layout.
// Block tile 128(M) x 128(N); one N-block == one entity (128 = 2*64).
// 8 waves; each wave: 2x4 WMMA 16x16 tiles (32x64 patch).
// Double-buffered LDS: stage(k+1) overlaps WMMA(k); 1 barrier / K-step.
// =====================================================================
__global__ __launch_bounds__(256) void dense_rope_kernel(
    const float* __restrict__ lhs,       // (4096,768)
    const float* __restrict__ cls_emb,   // (8,768)
    const float* __restrict__ dense_w,   // (6656,1536)
    const float* __restrict__ dense_b,   // (6656)
    __bf16* __restrict__ q_ws,           // (8,52,512,64) bf16
    __bf16* __restrict__ k_ws)           // (8,52,512,64) bf16
{
    // pad to 40 halfwords (80B): 16B-aligned rows, conflict-free (20 dwords)
    __shared__ __bf16 As[2][128][40];
    __shared__ __bf16 Bs[2][128][40];

    const int tid  = threadIdx.x;
    const int m0   = blockIdx.x * 128;
    const int ent  = blockIdx.y;          // N block == entity
    const int lane = tid & 31;
    const int wave = tid >> 5;
    const int hlf  = lane >> 4;           // 0: lanes 0-15, 1: lanes 16-31
    const int lrow = lane & 15;
    const int wm   = (wave & 3) * 32;     // wave M offset in tile
    const int wn   = (wave >> 2) * 64;    // wave N offset in tile

    const int ldRow = tid >> 1;           // 0..127
    const int ldCol = (tid & 1) * 16;     // 0 or 16

    const int   mld = m0 + ldRow;
    const int   bld = mld >> 9;           // /512
    const int   nld = ent * 128 + ldRow;

    // stage one 128x32 K-slab of A and B into buffer `buf` (float4 loads)
    auto stage = [&](int buf, int k0) {
        #pragma unroll
        for (int v = 0; v < 4; ++v) {
            const int kg = k0 + ldCol + v * 4;   // multiple of 4; 768 is mult of 16
            float4 f = (kg < HID)
                ? *(const float4*)(lhs + (size_t)mld * HID + kg)
                : *(const float4*)(cls_emb + (size_t)bld * HID + (kg - HID));
            As[buf][ldRow][ldCol + v * 4 + 0] = (__bf16)f.x;
            As[buf][ldRow][ldCol + v * 4 + 1] = (__bf16)f.y;
            As[buf][ldRow][ldCol + v * 4 + 2] = (__bf16)f.z;
            As[buf][ldRow][ldCol + v * 4 + 3] = (__bf16)f.w;
        }
        #pragma unroll
        for (int v = 0; v < 4; ++v) {
            const int kg = k0 + ldCol + v * 4;
            float4 f = *(const float4*)(dense_w + (size_t)nld * KDIM + kg);
            Bs[buf][ldRow][ldCol + v * 4 + 0] = (__bf16)f.x;
            Bs[buf][ldRow][ldCol + v * 4 + 1] = (__bf16)f.y;
            Bs[buf][ldRow][ldCol + v * 4 + 2] = (__bf16)f.z;
            Bs[buf][ldRow][ldCol + v * 4 + 3] = (__bf16)f.w;
        }
    };

    F32x8 acc[2][4];
    #pragma unroll
    for (int mt = 0; mt < 2; ++mt)
        #pragma unroll
        for (int nt = 0; nt < 4; ++nt) acc[mt][nt].v = (v8f){0,0,0,0,0,0,0,0};

    stage(0, 0);

    for (int kt = 0; kt < KDIM / 32; ++kt) {
        const int buf = kt & 1;
        __syncthreads();                       // stage(kt) visible to all
        if (kt + 1 < KDIM / 32) stage(buf ^ 1, (kt + 1) * 32);

        // ---- build fragments (ISA 7.12.2 16-bit A/B layouts) ----
        BF16x16 afrag[2], bfrag[4];
        #pragma unroll
        for (int mt = 0; mt < 2; ++mt)
            #pragma unroll
            for (int e = 0; e < 16; ++e) {
                const int kk = ((e & 8) << 1) + hlf * 8 + (e & 7);
                afrag[mt].e[e] = As[buf][wm + mt * 16 + lrow][kk];
            }
        #pragma unroll
        for (int nt = 0; nt < 4; ++nt)
            #pragma unroll
            for (int e = 0; e < 16; ++e) {
                const int kk = ((e & 8) << 1) + hlf * 8 + (e & 7);
                bfrag[nt].e[e] = Bs[buf][wn + nt * 16 + lrow][kk];
            }
        #pragma unroll
        for (int mt = 0; mt < 2; ++mt)
            #pragma unroll
            for (int nt = 0; nt < 4; ++nt)
                acc[mt][nt].v = __builtin_amdgcn_wmma_f32_16x16x32_bf16(
                    false, afrag[mt].v, false, bfrag[nt].v,
                    (short)0, acc[mt][nt].v, false, false);
    }

    // ---- epilogue: bias + RoPE, store bf16 q/k ----
    #pragma unroll
    for (int mt = 0; mt < 2; ++mt) {
        #pragma unroll
        for (int nt = 0; nt < 4; ++nt) {
            const int c     = wn + nt * 16 + lrow;        // 0..127 within entity
            const int nglob = ent * 128 + c;
            const float bias = dense_b[nglob];
            const int  isK = c >> 6;
            const int  d   = c & 63;
            const int  p   = d >> 1;
            const float invf = __expf(-LOG1E4_OVER_32 * (float)p);
            const float sgn  = (d & 1) ? 1.0f : -1.0f;
            __bf16* dst = isK ? k_ws : q_ws;
            #pragma unroll
            for (int r = 0; r < 8; ++r) {
                const int m = m0 + wm + mt * 16 + (hlf ? r + 8 : r);
                const int b = m >> 9;
                const int s = m & 511;
                float val = acc[mt][nt].e[r] + bias;
                // RoPE partner x[d^1] lives at lane^1 (N = lane%16)
                float partner = __shfl_xor(val, 1, 32);
                const float ang = (float)s * invf;
                const float cs = __cosf(ang);
                const float sn = __sinf(ang);
                const float out = val * cs + sgn * partner * sn;
                dst[(((size_t)b * NENT + ent) * SEQ + s) * DHEAD + d] = (__bf16)out;
            }
        }
    }
}

// =====================================================================
// Kernel C: per (b, ent) logits = q @ k^T, fused tril/pad mask + scale.
// grid (4 m-tiles, 4 n-tiles, 416 heads); block tile 128x128, K=64.
// bf16 tiles staged with CDNA5 async global->LDS (ASYNCcnt) when available.
// =====================================================================
__global__ __launch_bounds__(256) void logits_kernel(
    const __bf16* __restrict__ q_ws,
    const __bf16* __restrict__ k_ws,
    const float*  __restrict__ mask,     // (8,512)
    float* __restrict__ out)             // (8,52,512,512)
{
    // pad to 72 halfwords (144B): 16B-aligned rows for b128, stride 36 dwords
    __shared__ __bf16 Qs[128][72];
    __shared__ __bf16 Ks[128][72];

    const int tid  = threadIdx.x;
    const int z    = blockIdx.z;
    const int b    = z / NENT;
    const int h    = z % NENT;
    const int m0   = blockIdx.x * 128;
    const int n0   = blockIdx.y * 128;
    const int lane = tid & 31;
    const int wave = tid >> 5;
    const int hlf  = lane >> 4;
    const int lrow = lane & 15;
    const int wm   = (wave & 3) * 32;
    const int wn   = (wave >> 2) * 64;

    const __bf16* qbase = q_ws + ((size_t)b * NENT + h) * SEQ * DHEAD;
    const __bf16* kbase = k_ws + ((size_t)b * NENT + h) * SEQ * DHEAD;

    // stage tiles: row = tid/2 (0..127), cols (tid&1)*32 .. +31
    {
        const int r  = tid >> 1;
        const int c0 = (tid & 1) * 32;
#if HAVE_ASYNC_LDS
        #pragma unroll
        for (int v = 0; v < 4; ++v) {
            const int col = c0 + v * 8;      // 8 bf16 = 16B per async b128
            __builtin_amdgcn_global_load_async_to_lds_b128(
                (__attribute__((address_space(1))) v4i*)
                    (qbase + (size_t)(m0 + r) * DHEAD + col),
                (__attribute__((address_space(3))) v4i*)&Qs[r][col],
                0, 0);
            __builtin_amdgcn_global_load_async_to_lds_b128(
                (__attribute__((address_space(1))) v4i*)
                    (kbase + (size_t)(n0 + r) * DHEAD + col),
                (__attribute__((address_space(3))) v4i*)&Ks[r][col],
                0, 0);
        }
        wait_async_zero();
#else
        #pragma unroll
        for (int i = 0; i < 32; ++i)
            Qs[r][c0 + i] = qbase[(size_t)(m0 + r) * DHEAD + c0 + i];
        #pragma unroll
        for (int i = 0; i < 32; ++i)
            Ks[r][c0 + i] = kbase[(size_t)(n0 + r) * DHEAD + c0 + i];
#endif
    }
    __syncthreads();

    F32x8 acc[2][4];
    #pragma unroll
    for (int mt = 0; mt < 2; ++mt)
        #pragma unroll
        for (int nt = 0; nt < 4; ++nt) acc[mt][nt].v = (v8f){0,0,0,0,0,0,0,0};

    #pragma unroll
    for (int k0 = 0; k0 < DHEAD; k0 += 32) {
        BF16x16 afrag[2], bfrag[4];
        #pragma unroll
        for (int mt = 0; mt < 2; ++mt)
            #pragma unroll
            for (int e = 0; e < 16; ++e) {
                const int kk = k0 + ((e & 8) << 1) + hlf * 8 + (e & 7);
                afrag[mt].e[e] = Qs[wm + mt * 16 + lrow][kk];
            }
        #pragma unroll
        for (int nt = 0; nt < 4; ++nt)
            #pragma unroll
            for (int e = 0; e < 16; ++e) {
                const int kk = k0 + ((e & 8) << 1) + hlf * 8 + (e & 7);
                bfrag[nt].e[e] = Ks[wn + nt * 16 + lrow][kk];
            }
        #pragma unroll
        for (int mt = 0; mt < 2; ++mt)
            #pragma unroll
            for (int nt = 0; nt < 4; ++nt)
                acc[mt][nt].v = __builtin_amdgcn_wmma_f32_16x16x32_bf16(
                    false, afrag[mt].v, false, bfrag[nt].v,
                    (short)0, acc[mt][nt].v, false, false);
    }

    // ---- mask + scale epilogue, stream 436MB out ----
    float* obase = out + ((size_t)b * NENT + h) * SEQ * SEQ;
    #pragma unroll
    for (int mt = 0; mt < 2; ++mt) {
        #pragma unroll
        for (int nt = 0; nt < 4; ++nt) {
            const int n = n0 + wn + nt * 16 + lrow;
            const float pad = mask[(size_t)b * SEQ + n];
            #pragma unroll
            for (int r = 0; r < 8; ++r) {
                const int m = m0 + wm + mt * 16 + (hlf ? r + 8 : r);
                float val = acc[mt][nt].e[r];
                val = val * pad - (1.0f - pad) * BIGF;   // pad mask
                if (n < m) val -= BIGF;                  // tril(-1) causal mask
                obase[(size_t)m * SEQ + n] = val * 0.125f; // / sqrt(64)
            }
        }
    }
}

// =====================================================================
extern "C" void kernel_launch(void* const* d_in, const int* in_sizes, int n_in,
                              void* d_out, int out_size, void* d_ws, size_t ws_size,
                              hipStream_t stream) {
    const float* lhs        = (const float*)d_in[0];
    const float* cls_pooled = (const float*)d_in[1];
    const float* h0         = (const float*)d_in[2];
    const float* attn_mask  = (const float*)d_in[3];
    const float* w_ih_f     = (const float*)d_in[4];
    const float* w_hh_f     = (const float*)d_in[5];
    const float* b_ih_f     = (const float*)d_in[6];
    const float* b_hh_f     = (const float*)d_in[7];
    const float* w_ih_b     = (const float*)d_in[8];
    const float* w_hh_b     = (const float*)d_in[9];
    const float* b_ih_b     = (const float*)d_in[10];
    const float* b_hh_b     = (const float*)d_in[11];
    const float* dense_w    = (const float*)d_in[12];
    const float* dense_b    = (const float*)d_in[13];
    float* out = (float*)d_out;

    char* ws = (char*)d_ws;
    float*  cls_emb = (float*)ws;                                   // 24 KB
    const size_t qk_bytes = (size_t)NB * NENT * SEQ * DHEAD * 2;    // ~26 MB each
    __bf16* q_ws = (__bf16*)(ws + 32768);
    __bf16* k_ws = (__bf16*)(ws + 32768 + qk_bytes);

    // A: GRU cells -> cls_emb
    gru_kernel<<<24, 256, 0, stream>>>(cls_pooled, h0,
        w_ih_f, w_hh_f, b_ih_f, b_hh_f,
        w_ih_b, w_hh_b, b_ih_b, b_hh_b, cls_emb);

    // B: big WMMA GEMM + bias + RoPE -> bf16 q/k (double-buffered)
    dense_rope_kernel<<<dim3(32, 52, 1), 256, 0, stream>>>(
        lhs, cls_emb, dense_w, dense_b, q_ws, k_ws);

    // C: per-head WMMA logits + masks + scale (async LDS staging)
    logits_kernel<<<dim3(4, 4, 416), 256, 0, stream>>>(
        q_ws, k_ws, attn_mask, out);
}
